// GraphUNet_22634477649991
// MI455X (gfx1250) — matrix operand order repaired and checked by reading, not compiled, not run on hardware
//
#include <hip/hip_runtime.h>
#include <hip/hip_bf16.h>

typedef __attribute__((ext_vector_type(16))) _Float16 v16h;
typedef __attribute__((ext_vector_type(8)))  _Float16 v8h;
typedef __attribute__((ext_vector_type(4)))  _Float16 h4;
typedef __attribute__((ext_vector_type(8)))  float    v8f;
typedef __attribute__((ext_vector_type(4)))  float    f4;

// ---------------------------------------------------------------------------
// WMMA GEMM: C(MxN) = A(MxK) @ B(KxN) [+bias][ReLU], f32 in memory,
// f16 operands / f32 accumulate via v_wmma_f32_16x16x32_f16.
// Block tile 128x128x64, 256 threads = 8 waves (2x4), wave tile 64x32.
// Both LDS stages are K-contiguous so every fragment load is 2x ds_load_b128.
// Requires M,N multiples of 128 and K multiple of 64 (true for every call).
// ---------------------------------------------------------------------------
#define BM 128
#define BN 128
#define BK 64
#define KPAD 8   // row stride BK+8 = 72 halves = 144B (16B aligned, odd dword skew)

__global__ __launch_bounds__(256)
void wmma_gemm_kernel(const float* __restrict__ A, const float* __restrict__ B,
                      float* __restrict__ C, const float* __restrict__ bias,
                      int M, int N, int K, int relu) {
  __shared__ _Float16 As[BM][BK + KPAD];  // M-major, K contiguous
  __shared__ _Float16 Bt[BN][BK + KPAD];  // N-major, K contiguous (transposed)

  const int tid  = threadIdx.x;
  const int lane = tid & 31;
  const int wid  = tid >> 5;        // 0..7
  const int rowW = (wid >> 2) * 64; // 2 waves in M
  const int colW = (wid & 3) * 32;  // 4 waves in N
  const int blockRow = blockIdx.y * BM;
  const int blockCol = blockIdx.x * BN;

  // ISA wave32 fragment layouts (CDNA5 ISA 7.12.2):
  const int mA    = lane & 15;
  const int kAoff = (lane >> 4) << 3;  // A: lanes16-31 take K sub-block +8
  const int nB    = lane & 15;
  const int kBoff = (lane >> 4) << 3;  // B: lanes16-31 take K sub-block +8
  const int mCoff = (lane >> 4) << 3;  // C/D: VGPR i -> M=i / M=8+i
  const int nC    = lane & 15;

  v8f vzero = {};
  v8f acc[4][2];
  for (int i = 0; i < 4; ++i)
    for (int j = 0; j < 2; ++j) acc[i][j] = vzero;

  for (int kt = 0; kt < K; kt += BK) {
    // Stage A: 128x64, float4 global loads, packed h4 LDS stores (conflict-free)
    for (int idx = tid; idx < BM * (BK / 4); idx += 256) {
      int r = idx >> 4;            // 16 float4-chunks per row
      int c = (idx & 15) << 2;
      f4 v = *(const f4*)&A[(size_t)(blockRow + r) * K + kt + c];
      h4 h = {(_Float16)v.x, (_Float16)v.y, (_Float16)v.z, (_Float16)v.w};
      *(h4*)&As[r][c] = h;
    }
    // Stage B transposed: per thread, 4 coalesced b32 loads down a column group,
    // one packed h4 store into the N-major tile.
    for (int idx = tid; idx < (BK / 4) * BN; idx += 256) {
      int n  = idx & 127;          // fastest across lanes -> coalesced global
      int kc = (idx >> 7) << 2;
      h4 h;
      for (int j = 0; j < 4; ++j)
        h[j] = (_Float16)B[(size_t)(kt + kc + j) * N + blockCol + n];
      *(h4*)&Bt[n][kc] = h;
    }
    __syncthreads();

    for (int ks = 0; ks < BK; ks += 32) {
      v16h afrag[4];
#pragma unroll
      for (int mi = 0; mi < 4; ++mi) {
        const _Float16* ap = &As[rowW + mi * 16 + mA][ks];
        v8h lo = *(const v8h*)(ap + kAoff);        // K ks+kAoff .. +7
        v8h hi = *(const v8h*)(ap + 16 + kAoff);   // K ks+16+kAoff .. +7
        afrag[mi] = __builtin_shufflevector(lo, hi,
            0, 1, 2, 3, 4, 5, 6, 7, 8, 9, 10, 11, 12, 13, 14, 15);
      }
#pragma unroll
      for (int ni = 0; ni < 2; ++ni) {
        const _Float16* bp = &Bt[colW + ni * 16 + nB][ks + kBoff];
        v8h lo = *(const v8h*)(bp);                // K ks+kBoff .. +7
        v8h hi = *(const v8h*)(bp + 16);           // K ks+16+kBoff .. +7
        v16h bfrag = __builtin_shufflevector(lo, hi,
            0, 1, 2, 3, 4, 5, 6, 7, 8, 9, 10, 11, 12, 13, 14, 15);
#pragma unroll
        for (int mi = 0; mi < 4; ++mi) {
          acc[mi][ni] = __builtin_amdgcn_wmma_f32_16x16x32_f16(
              false, afrag[mi], false, bfrag, (short)0, acc[mi][ni], false, false);
        }
      }
    }
    __syncthreads();
  }

  // Epilogue. C/D layout: VGPR i -> M=i (lanes 0-15) / M=8+i (lanes 16-31)
  for (int mi = 0; mi < 4; ++mi) {
    for (int ni = 0; ni < 2; ++ni) {
      int r0 = blockRow + rowW + mi * 16 + mCoff;
      int c  = blockCol + colW + ni * 16 + nC;
      float bv = bias ? bias[c] : 0.0f;
#pragma unroll
      for (int i = 0; i < 8; ++i) {
        float v = acc[mi][ni][i] + bv;
        if (relu) v = fmaxf(v, 0.0f);
        C[(size_t)(r0 + i) * N + c] = v;
      }
    }
  }
}

// ---------------------------------------------------------------------------
// Supporting kernels
// ---------------------------------------------------------------------------
__global__ void zero_kernel(float* p, size_t count) {
  size_t i = (size_t)blockIdx.x * blockDim.x + threadIdx.x;
  if (i < count) p[i] = 0.0f;
}

__global__ void copy_kernel(const float* __restrict__ in, float* __restrict__ out,
                            size_t count) {
  size_t i = (size_t)blockIdx.x * blockDim.x + threadIdx.x;
  if (i < count) out[i] = in[i];
}

__global__ void edges_kernel(const int* __restrict__ ei, float* __restrict__ M,
                             int E, int n) {
  int e = blockIdx.x * blockDim.x + threadIdx.x;
  if (e < E) {
    int src = ei[e], dst = ei[E + e];
    atomicAdd(&M[(size_t)src * n + dst], 1.0f);
  }
}

// dis[c] = rsqrt(1 + colsum(M)[:,c])   (self-loop included, deg >= 1 always)
__global__ void dis_kernel(const float* __restrict__ M, float* __restrict__ dis, int n) {
  int c = blockIdx.x * blockDim.x + threadIdx.x;
  if (c >= n) return;
  float s = 1.0f;
  for (int r = 0; r < n; ++r) s += M[(size_t)r * n + c];
  dis[c] = rsqrtf(s);
}

// An[d][s] = dis[d]*dis[s]*(M[s][d] + (s==d))   — normalized A transposed,
// LDS-tiled so both global read and write are coalesced.
__global__ void norm_adj_kernel(const float* __restrict__ M, const float* __restrict__ dis,
                                float* __restrict__ An, int n) {
  __shared__ float t[32][33];
  int d0 = blockIdx.x * 32, s0 = blockIdx.y * 32;
  int tx = threadIdx.x, ty = threadIdx.y;  // 32 x 8
  for (int j = 0; j < 32; j += 8)
    t[ty + j][tx] = M[(size_t)(s0 + ty + j) * n + d0 + tx];
  __syncthreads();
  for (int j = 0; j < 32; j += 8) {
    int d = d0 + ty + j, s = s0 + tx;
    float v = t[tx][ty + j] + (d == s ? 1.0f : 0.0f);
    An[(size_t)d * n + s] = dis[d] * dis[s] * v;
  }
}

__global__ void addI_kernel(const float* __restrict__ M, float* __restrict__ B, int n) {
  size_t i = (size_t)blockIdx.x * blockDim.x + threadIdx.x;
  size_t total = (size_t)n * n;
  if (i >= total) return;
  int r = (int)(i / n), c = (int)(i % n);
  B[i] = M[i] + (r == c ? 1.0f : 0.0f);
}

__global__ void pnorm_kernel(const float* __restrict__ p, int f, float* __restrict__ out) {
  __shared__ float red[256];
  float s = 0.0f;
  for (int i = threadIdx.x; i < f; i += 256) s += p[i] * p[i];
  red[threadIdx.x] = s;
  __syncthreads();
  for (int st = 128; st > 0; st >>= 1) {
    if ((int)threadIdx.x < st) red[threadIdx.x] += red[threadIdx.x + st];
    __syncthreads();
  }
  if (threadIdx.x == 0) out[0] = rsqrtf(red[0]);
}

// score[i] = tanh(dot(x[i], p) * pinv)   — one wave32 per node
__global__ void score_kernel(const float* __restrict__ x, const float* __restrict__ p,
                             const float* __restrict__ pinv, float* __restrict__ out,
                             int n, int f) {
  int node = blockIdx.x * (blockDim.x >> 5) + (threadIdx.x >> 5);
  int lane = threadIdx.x & 31;
  if (node >= n) return;
  float sum = 0.0f;
  for (int c = lane; c < f; c += 32) sum += x[(size_t)node * f + c] * p[c];
  for (int off = 16; off > 0; off >>= 1) sum += __shfl_down(sum, off, 32);
  if (lane == 0) out[node] = tanhf(sum * pinv[0]);
}

// Single-workgroup bitonic sort (descending) of (score, index); emit top-k.
__global__ __launch_bounds__(1024)
void topk_kernel(const float* __restrict__ score, int n, int k,
                 int* __restrict__ perm, float* __restrict__ vals) {
  __shared__ float s[4096];
  __shared__ int   id[4096];
  for (int i = threadIdx.x; i < n; i += blockDim.x) { s[i] = score[i]; id[i] = i; }
  __syncthreads();
  for (int len = 2; len <= n; len <<= 1) {
    for (int j = len >> 1; j > 0; j >>= 1) {
      for (int i = threadIdx.x; i < n; i += blockDim.x) {
        int ixj = i ^ j;
        if (ixj > i) {
          bool desc = ((i & len) == 0);
          float a = s[i], b = s[ixj];
          bool sw = desc ? (a < b) : (a > b);
          if (sw) {
            s[i] = b; s[ixj] = a;
            int t = id[i]; id[i] = id[ixj]; id[ixj] = t;
          }
        }
      }
      __syncthreads();
    }
  }
  for (int i = threadIdx.x; i < k; i += blockDim.x) { perm[i] = id[i]; vals[i] = s[i]; }
}

__global__ void gather_x_kernel(const float* __restrict__ x, const int* __restrict__ perm,
                                const float* __restrict__ vals, float* __restrict__ xp,
                                int f, int k) {
  size_t i = (size_t)blockIdx.x * blockDim.x + threadIdx.x;
  size_t total = (size_t)k * f;
  if (i >= total) return;
  int r = (int)(i / f), c = (int)(i % f);
  xp[i] = x[(size_t)perm[r] * f + c] * vals[r];
}

__global__ void gather_adj_kernel(const float* __restrict__ Aug, const int* __restrict__ perm,
                                  float* __restrict__ Mp, int n, int k) {
  size_t i = (size_t)blockIdx.x * blockDim.x + threadIdx.x;
  size_t total = (size_t)k * k;
  if (i >= total) return;
  int r = (int)(i / k), c = (int)(i % k);
  Mp[i] = Aug[(size_t)perm[r] * n + perm[c]];
}

__global__ void scatter_add_kernel(const float* __restrict__ xs, const int* __restrict__ perm,
                                   float* __restrict__ xin, int f, int k) {
  size_t i = (size_t)blockIdx.x * blockDim.x + threadIdx.x;
  size_t total = (size_t)k * f;
  if (i >= total) return;
  int r = (int)(i / f), c = (int)(i % f);
  xin[(size_t)perm[r] * f + c] += xs[i];
}

// ---------------------------------------------------------------------------
// Host-side orchestration
// ---------------------------------------------------------------------------
static inline size_t cdiv(size_t a, size_t b) { return (a + b - 1) / b; }

static void launch_gemm(const float* A, const float* B, float* C, const float* bias,
                        int M, int N, int K, int relu, hipStream_t stream) {
  dim3 grid(N / 128, M / 128);
  wmma_gemm_kernel<<<grid, dim3(256), 0, stream>>>(A, B, C, bias, M, N, K, relu);
}

// GCN: out = relu( An @ (x @ w) + b ),  An = normalized-transposed (M + I)
static void gcn_step(const float* Madj, const float* x, const float* w, const float* b,
                     float* An, float* xw, float* out, float* dis,
                     int n, int fi, int fo, hipStream_t stream) {
  dis_kernel<<<dim3((unsigned)cdiv(n, 256)), dim3(256), 0, stream>>>(Madj, dis, n);
  norm_adj_kernel<<<dim3(n / 32, n / 32), dim3(32, 8), 0, stream>>>(Madj, dis, An, n);
  launch_gemm(x, w, xw, nullptr, n, fo, fi, 0, stream);
  launch_gemm(An, xw, out, b, n, fo, n, 1, stream);
}

static void pool_step(const float* xfeat, const float* p, const float* Aug,
                      int n, int f, int k, int* perm, float* vals,
                      float* xp, float* Mp, float* scores, float* pnrm,
                      hipStream_t stream) {
  pnorm_kernel<<<1, 256, 0, stream>>>(p, f, pnrm);
  score_kernel<<<dim3(n / 8), dim3(256), 0, stream>>>(xfeat, p, pnrm, scores, n, f);
  topk_kernel<<<1, 1024, 0, stream>>>(scores, n, k, perm, vals);
  size_t tx = (size_t)k * f;
  gather_x_kernel<<<dim3((unsigned)cdiv(tx, 256)), dim3(256), 0, stream>>>(xfeat, perm, vals, xp, f, k);
  size_t ta = (size_t)k * k;
  gather_adj_kernel<<<dim3((unsigned)cdiv(ta, 256)), dim3(256), 0, stream>>>(Aug, perm, Mp, n, k);
}

extern "C" void kernel_launch(void* const* d_in, const int* in_sizes, int n_in,
                              void* d_out, int out_size, void* d_ws, size_t ws_size,
                              hipStream_t stream) {
  (void)in_sizes; (void)n_in; (void)out_size; (void)ws_size;

  const int N0 = 4096, N1 = 2048, N2 = 1024, E = 131072;

  const float* x  = (const float*)d_in[0];
  const int*   ei = (const int*)d_in[1];
  const float* w0 = (const float*)d_in[2];  const float* b0 = (const float*)d_in[3];
  const float* w1 = (const float*)d_in[4];  const float* b1 = (const float*)d_in[5];
  const float* w2 = (const float*)d_in[6];  const float* b2 = (const float*)d_in[7];
  const float* w3 = (const float*)d_in[8];  const float* b3 = (const float*)d_in[9];
  const float* w4 = (const float*)d_in[10]; const float* b4 = (const float*)d_in[11];
  const float* p0 = (const float*)d_in[12];
  const float* p1 = (const float*)d_in[13];
  float* out = (float*)d_out;

  // Workspace layout (every byte read is written first; harness poisons 0xAA)
  char* ws = (char*)d_ws;
  size_t off = 0;
  auto alloc = [&](size_t bytes) -> char* {
    char* p = ws + off;
    off += (bytes + 255) & ~(size_t)255;
    return p;
  };
  float* M0   = (float*)alloc((size_t)N0 * N0 * 4);   // adjacency, persistent
  float* BIG1 = (float*)alloc((size_t)N0 * N0 * 4);   // An / (M+I) scratch
  float* BIG2 = (float*)alloc((size_t)N0 * N0 * 4);   // augment output
  float* M1   = (float*)alloc((size_t)N1 * N1 * 4);
  float* M2   = (float*)alloc((size_t)N2 * N2 * 4);
  float* XW   = (float*)alloc((size_t)N0 * 512 * 4);  // x@w temp
  float* RES0 = (float*)alloc((size_t)N0 * 256 * 4);
  float* XB   = (float*)alloc((size_t)N1 * 256 * 4);
  float* RES1 = (float*)alloc((size_t)N1 * 512 * 4);
  float* XD   = (float*)alloc((size_t)N2 * 512 * 4);
  float* XE   = (float*)alloc((size_t)N2 * 512 * 4);
  float* XIN1 = (float*)alloc((size_t)N1 * 512 * 4);
  float* XF   = (float*)alloc((size_t)N1 * 256 * 4);
  float* XIN0 = (float*)alloc((size_t)N0 * 256 * 4);
  float* DIS  = (float*)alloc((size_t)N0 * 4);
  float* SCR  = (float*)alloc((size_t)N0 * 4);
  float* PNRM = (float*)alloc(256);
  int*   PERM0 = (int*)alloc((size_t)N1 * 4);
  float* VALS0 = (float*)alloc((size_t)N1 * 4);
  int*   PERM1 = (int*)alloc((size_t)N2 * 4);
  float* VALS1 = (float*)alloc((size_t)N2 * 4);

  // 1. Dense adjacency M0[src][dst] from edge list
  size_t m0elems = (size_t)N0 * N0;
  zero_kernel<<<dim3((unsigned)cdiv(m0elems, 256)), dim3(256), 0, stream>>>(M0, m0elems);
  edges_kernel<<<dim3((unsigned)cdiv(E, 256)), dim3(256), 0, stream>>>(ei, M0, E, N0);

  // 2. Down block 0: RES0 = GCN(x, M0, w0, b0)        [4096 x 256]
  gcn_step(M0, x, w0, b0, BIG1, XW, RES0, DIS, N0, 128, 256, stream);

  // 3. Augment 0: BIG2 = (M0+I)@(M0+I)                [4096^3 — dominant GEMM]
  addI_kernel<<<dim3((unsigned)cdiv(m0elems, 256)), dim3(256), 0, stream>>>(M0, BIG1, N0);
  launch_gemm(BIG1, BIG1, BIG2, nullptr, N0, N0, N0, 0, stream);

  // 4. Pool 0 -> XB [2048x256], M1 [2048x2048], PERM0/VALS0
  pool_step(RES0, p0, BIG2, N0, 256, N1, PERM0, VALS0, XB, M1, SCR, PNRM, stream);

  // 5. Down block 1: RES1 = GCN(XB, M1, w1, b1)       [2048 x 512]
  gcn_step(M1, XB, w1, b1, BIG1, XW, RES1, DIS, N1, 256, 512, stream);

  // 6. Augment 1: BIG2 = (M1+I)@(M1+I)
  size_t m1elems = (size_t)N1 * N1;
  addI_kernel<<<dim3((unsigned)cdiv(m1elems, 256)), dim3(256), 0, stream>>>(M1, BIG1, N1);
  launch_gemm(BIG1, BIG1, BIG2, nullptr, N1, N1, N1, 0, stream);

  // 7. Pool 1 -> XD [1024x512], M2 [1024x1024], PERM1/VALS1
  pool_step(RES1, p1, BIG2, N1, 512, N2, PERM1, VALS1, XD, M2, SCR, PNRM, stream);

  // 8. Bottleneck: XE = GCN(XD, M2, w2, b2)           [1024 x 512]
  gcn_step(M2, XD, w2, b2, BIG1, XW, XE, DIS, N2, 512, 512, stream);

  // 9. Up block 1: XIN1 = RES1 + scatter(PERM1, XE);  XF = GCN(XIN1, M1, w3, b3)
  size_t r1elems = (size_t)N1 * 512;
  copy_kernel<<<dim3((unsigned)cdiv(r1elems, 256)), dim3(256), 0, stream>>>(RES1, XIN1, r1elems);
  scatter_add_kernel<<<dim3((unsigned)cdiv((size_t)N2 * 512, 256)), dim3(256), 0, stream>>>(
      XE, PERM1, XIN1, 512, N2);
  gcn_step(M1, XIN1, w3, b3, BIG1, XW, XF, DIS, N1, 512, 256, stream);

  // 10. Up block 0: XIN0 = RES0 + scatter(PERM0, XF); out = GCN(XIN0, M0, w4, b4)
  size_t r0elems = (size_t)N0 * 256;
  copy_kernel<<<dim3((unsigned)cdiv(r0elems, 256)), dim3(256), 0, stream>>>(RES0, XIN0, r0elems);
  scatter_add_kernel<<<dim3((unsigned)cdiv((size_t)N1 * 256, 256)), dim3(256), 0, stream>>>(
      XF, PERM0, XIN0, 256, N1);
  gcn_step(M0, XIN0, w4, b4, BIG1, XW, out, DIS, N0, 256, 128, stream);
}
